// AutoregressiveDecoder_82712480186644
// MI455X (gfx1250) — matrix operand order repaired and checked by reading, not compiled, not run on hardware
//
#include <hip/hip_runtime.h>

// ---------------------------------------------------------------------------
// Autoregressive 2-layer GRU decoder for MI455X (gfx1250, wave32, WMMA bf16)
//   B=32768, C=512, H=256, 3H=768, pred_len=60 (runtime from d_in[1])
//   Round 2: gi0 tile + projection weights LDS-resident for the whole 60-step
//   loop; all time-invariant scalars hoisted; LDS stride tuned (264 shorts
//   == 4 dwords mod 64 banks -> optimal 2-way pattern for b128 A-frag loads).
// ---------------------------------------------------------------------------

#define H_    256
#define G3    768
#define CDIM  512
#define ROWS  32
#define LDSW  264   // h-state row stride in bf16 (132 dwords = 4 mod 64 banks)
#define GIW   772   // gi0 row stride in floats  (8*772 = 32 mod 64 banks)

typedef __bf16 v16bf __attribute__((ext_vector_type(16)));
typedef float  v8f   __attribute__((ext_vector_type(8)));

union FragA { uint4 q[2]; v16bf v; };
union FragC { v8f v; float f[8]; };

#define WMMA(a, b, c) \
  __builtin_amdgcn_wmma_f32_16x16x32_bf16(false, (a), false, (b), (short)0, (c), false, false)

__device__ __forceinline__ unsigned short f2bf(float f) {
  unsigned u = __float_as_uint(f);
  u += 0x7FFFu + ((u >> 16) & 1u);          // round-to-nearest-even
  return (unsigned short)(u >> 16);
}
__device__ __forceinline__ float bflo(unsigned v) { return __uint_as_float(v << 16); }
__device__ __forceinline__ float bfhi(unsigned v) { return __uint_as_float(v & 0xFFFF0000u); }
__device__ __forceinline__ float sigmoidf_(float x) { return 1.f / (1.f + __expf(-x)); }
__device__ __forceinline__ float softplusf_(float x) { return (x > 20.f) ? x : log1pf(__expf(x)); }

// ---------------- fp32 -> bf16 converters -----------------------------------
__global__ void cvt_bf16(const float* __restrict__ s, unsigned short* __restrict__ d, int n) {
  for (int i = blockIdx.x * blockDim.x + threadIdx.x; i < n; i += gridDim.x * blockDim.x)
    d[i] = f2bf(s[i]);
}
// strided copy: drop the first `off` columns (w_ih0 pos columns stay fp32)
__global__ void cvt_bf16_strided(const float* __restrict__ s, unsigned short* __restrict__ d,
                                 int rows, int cols, int sstride, int off) {
  int n = rows * cols;
  for (int i = blockIdx.x * blockDim.x + threadIdx.x; i < n; i += gridDim.x * blockDim.x) {
    int r = i / cols, c = i - r * cols;
    d[i] = f2bf(s[(size_t)r * sstride + off + c]);
  }
}

// ---------------- init GEMM -------------------------------------------------
// hid = tanh(ctx@Winit^T + b_init) -> h0,h1 ; gi0 = ctx@Wih0c^T + b_ih0 (+b_hh0
// folded for the r,z gate columns only; the n-gate b_hh0 lives under r*( ) and
// is added in the decoder). grid = (B/64, 1280/64), 8 waves of 16x32.
__global__ __launch_bounds__(256) void init_gemm(
    const unsigned short* __restrict__ ctxb,    // B x 512 bf16
    const unsigned short* __restrict__ winitb,  // 512 x 512 bf16
    const unsigned short* __restrict__ wih0cb,  // 768 x 512 bf16
    const float* __restrict__ b_init, const float* __restrict__ b_ih0,
    const float* __restrict__ b_hh0,
    float* __restrict__ h0f, float* __restrict__ h1f, float* __restrict__ gi0) {
  const int lane = threadIdx.x & 31, wave = threadIdx.x >> 5;
  const int wm = wave & 3, wn = wave >> 2;
  const int laneLo = lane & 15, laneHi = lane >> 4;
  const int rowA = blockIdx.x * 64 + wm * 16 + laneLo;
  const int colBase = blockIdx.y * 64 + wn * 32;
  const unsigned short* wsrc = (colBase < CDIM) ? winitb : wih0cb;

  FragC acc[2];
#pragma unroll
  for (int nt = 0; nt < 2; ++nt)
#pragma unroll
    for (int i = 0; i < 8; ++i) acc[nt].f[i] = 0.f;

  for (int kk = 0; kk < 16; ++kk) {
    const int koff = kk * 32;
    FragA a;
    const unsigned short* ap = ctxb + (size_t)rowA * CDIM + koff + laneHi * 16;
    a.q[0] = *(const uint4*)ap;
    a.q[1] = *(const uint4*)(ap + 8);
#pragma unroll
    for (int nt = 0; nt < 2; ++nt) {
      const int ncol = colBase + nt * 16 + laneLo;
      const int nl = (ncol < CDIM) ? ncol : (ncol - CDIM);
      FragA b;
      const unsigned short* bp = wsrc + (size_t)nl * CDIM + koff + laneHi * 16;
      b.q[0] = *(const uint4*)bp;
      b.q[1] = *(const uint4*)(bp + 8);
      acc[nt].v = WMMA(a.v, b.v, acc[nt].v);
    }
  }
#pragma unroll
  for (int nt = 0; nt < 2; ++nt) {
    const int col = colBase + nt * 16 + laneLo;
#pragma unroll
    for (int r = 0; r < 8; ++r) {
      const int row = blockIdx.x * 64 + wm * 16 + r + 8 * laneHi;
      float v = acc[nt].f[r];
      if (col < CDIM) {
        float h = tanhf(v + b_init[col]);
        if (col < H_) h0f[(size_t)row * H_ + col] = h;
        else          h1f[(size_t)row * H_ + (col - H_)] = h;
      } else {
        const int jc = col - CDIM;
        float bias = b_ih0[jc] + ((jc < 2 * H_) ? b_hh0[jc] : 0.f);  // fold r,z
        gi0[(size_t)row * G3 + jc] = v + bias;
      }
    }
  }
}

// ---------------- persistent decoder ----------------------------------------
#define LOAD_A(frag, hs, mt, koff)                                                   \
  do {                                                                               \
    const unsigned short* _p = &hs[((mt)*16 + laneLo) * LDSW + (koff) + laneHi * 16]; \
    (frag).q[0] = *(const uint4*)_p;                                                 \
    (frag).q[1] = *(const uint4*)(_p + 8);                                           \
  } while (0)
#define LOAD_B(frag, wb, j, koff)                                                    \
  do {                                                                               \
    const unsigned short* _p = (wb) + (size_t)(j)*H_ + (koff) + laneHi * 16;         \
    (frag).q[0] = *(const uint4*)_p;                                                 \
    (frag).q[1] = *(const uint4*)(_p + 8);                                           \
  } while (0)

__global__ __launch_bounds__(256) void decode(
    const unsigned short* __restrict__ whh0b,   // 768 x 256 bf16
    const unsigned short* __restrict__ wih1b,   // 768 x 256 bf16
    const unsigned short* __restrict__ whh1b,   // 768 x 256 bf16
    const float* __restrict__ gi0,              // B x 768 (b_ih0 [+b_hh0 rz] folded)
    const float* __restrict__ wih0,             // 768 x 514 fp32 (pos cols 0..1)
    const float* __restrict__ b_hh0, const float* __restrict__ b_ih1,
    const float* __restrict__ b_hh1,
    const float* __restrict__ h0f, const float* __restrict__ h1f,
    const float* __restrict__ wpos, const float* __restrict__ bpos,
    const float* __restrict__ wchol, const float* __restrict__ bchol,
    const int* __restrict__ predlen, float* __restrict__ out, int Btot) {
  __shared__ __align__(16) float gis[ROWS * GIW];            // 96.5 KB
  __shared__ __align__(16) unsigned short h0s[ROWS * LDSW];  // 16.5 KB
  __shared__ __align__(16) unsigned short h1s[ROWS * LDSW];  // 16.5 KB
  __shared__ __align__(16) float wps[5 * H_];                // 5 KB
  __shared__ float poss[ROWS * 2];

  const int tid = threadIdx.x;
  const int lane = tid & 31, wave = tid >> 5;
  const int laneLo = lane & 15, laneHi = lane >> 4;
  const int row0 = blockIdx.x * ROWS;
  const int PL = predlen[0];
  const size_t outCh = (size_t)Btot * PL * 2;

  // ---- one-time LDS seeding --------------------------------------------
  for (int i = tid; i < ROWS * H_; i += 256) {
    const int r = i >> 8, c = i & (H_ - 1);
    h0s[r * LDSW + c] = f2bf(h0f[(size_t)(row0 + r) * H_ + c]);
    h1s[r * LDSW + c] = f2bf(h1f[(size_t)(row0 + r) * H_ + c]);
  }
  for (int i = tid; i < ROWS * G3; i += 256) {
    const int r = i / G3, c = i - r * G3;
    gis[r * GIW + c] = gi0[(size_t)(row0 + r) * G3 + c];
  }
  for (int i = tid; i < 5 * H_; i += 256)
    wps[i] = (i < 2 * H_) ? wpos[i] : wchol[i - 2 * H_];
  if (tid < ROWS * 2) poss[tid] = 0.f;

  // ---- time-invariant per-wave scalars ---------------------------------
  float wr0[2], wr1[2], wz0[2], wz1[2], wn0[2], wn1[2];   // w_ih0 pos columns
  float bn0[2];                                           // layer0 n-gate b_hh0
  float br1[2], bz1[2], bin1[2], bhn1[2];                 // layer1 biases
#pragma unroll
  for (int nt = 0; nt < 2; ++nt) {
    const int jc = wave * 32 + nt * 16 + laneLo;
    wr0[nt] = wih0[(size_t)jc * 514];            wr1[nt] = wih0[(size_t)jc * 514 + 1];
    wz0[nt] = wih0[(size_t)(jc + H_) * 514];     wz1[nt] = wih0[(size_t)(jc + H_) * 514 + 1];
    wn0[nt] = wih0[(size_t)(jc + 2 * H_) * 514]; wn1[nt] = wih0[(size_t)(jc + 2 * H_) * 514 + 1];
    bn0[nt] = b_hh0[jc + 2 * H_];
    br1[nt] = b_ih1[jc] + b_hh1[jc];
    bz1[nt] = b_ih1[jc + H_] + b_hh1[jc + H_];
    bin1[nt] = b_ih1[jc + 2 * H_];
    bhn1[nt] = b_hh1[jc + 2 * H_];
  }
  // projection task constants
  int prow = 0, po = 0;
  float pbias = 0.f;
  if (tid < ROWS * 5) {
    prow = tid / 5; po = tid - prow * 5;
    pbias = (po < 2) ? bpos[po] : bchol[po - 2];
  }

  // fp32 register copies of this wave's 32 columns (same mapping as D-frag)
  FragC h0r[2][2], h1r[2][2];
#pragma unroll
  for (int nt = 0; nt < 2; ++nt)
#pragma unroll
    for (int mt = 0; mt < 2; ++mt)
#pragma unroll
      for (int r = 0; r < 8; ++r) {
        const int m = mt * 16 + r + 8 * laneHi;
        const int c = wave * 32 + nt * 16 + laneLo;
        h0r[nt][mt].f[r] = h0f[(size_t)(row0 + m) * H_ + c];
        h1r[nt][mt].f[r] = h1f[(size_t)(row0 + m) * H_ + c];
      }
  __syncthreads();

  for (int t = 0; t < PL; ++t) {
    // ------- layer 0: gh0 = h0 @ w_hh0^T (WMMA) -------
    FragC acc[3][2][2];
#pragma unroll
    for (int g = 0; g < 3; ++g)
#pragma unroll
      for (int nt = 0; nt < 2; ++nt)
#pragma unroll
        for (int mt = 0; mt < 2; ++mt)
#pragma unroll
          for (int i = 0; i < 8; ++i) acc[g][nt][mt].f[i] = 0.f;

    for (int kk = 0; kk < 8; ++kk) {
      const int koff = kk * 32;
      FragA a0, a1;
      LOAD_A(a0, h0s, 0, koff);
      LOAD_A(a1, h0s, 1, koff);
#pragma unroll
      for (int g = 0; g < 3; ++g)
#pragma unroll
        for (int nt = 0; nt < 2; ++nt) {
          const int j = g * H_ + wave * 32 + nt * 16 + laneLo;
          FragA b;
          LOAD_B(b, whh0b, j, koff);
          acc[g][nt][0].v = WMMA(a0.v, b.v, acc[g][nt][0].v);
          acc[g][nt][1].v = WMMA(a1.v, b.v, acc[g][nt][1].v);
        }
    }
    // combine gates -> new h0 (registers); gi from LDS, pos rank-2 update
#pragma unroll
    for (int mt = 0; mt < 2; ++mt)
#pragma unroll
      for (int r = 0; r < 8; ++r) {
        const int m = mt * 16 + r + 8 * laneHi;
        const float p0 = poss[m * 2], p1 = poss[m * 2 + 1];
#pragma unroll
        for (int nt = 0; nt < 2; ++nt) {
          const int jc = wave * 32 + nt * 16 + laneLo;
          const float* gp = &gis[m * GIW + jc];
          const float ir = gp[0] + p0 * wr0[nt] + p1 * wr1[nt];
          const float iz = gp[H_] + p0 * wz0[nt] + p1 * wz1[nt];
          const float in_ = gp[2 * H_] + p0 * wn0[nt] + p1 * wn1[nt];
          const float rg = sigmoidf_(ir + acc[0][nt][mt].f[r]);
          const float zg = sigmoidf_(iz + acc[1][nt][mt].f[r]);
          const float ng = tanhf(in_ + rg * (acc[2][nt][mt].f[r] + bn0[nt]));
          h0r[nt][mt].f[r] = (1.f - zg) * ng + zg * h0r[nt][mt].f[r];
        }
      }
    __syncthreads();  // all waves finished reading old h0s
#pragma unroll
    for (int nt = 0; nt < 2; ++nt)
#pragma unroll
      for (int mt = 0; mt < 2; ++mt)
#pragma unroll
        for (int r = 0; r < 8; ++r) {
          const int m = mt * 16 + r + 8 * laneHi;
          h0s[m * LDSW + wave * 32 + nt * 16 + laneLo] = f2bf(h0r[nt][mt].f[r]);
        }
    __syncthreads();  // new h0s visible

    // ------- layer 1: rz = [h0;h1]@[w_ih1;w_hh1]^T (r,z joint K=512) --------
    FragC rz[2][2][2], inn[2][2], hnn[2][2];
#pragma unroll
    for (int g = 0; g < 2; ++g)
#pragma unroll
      for (int nt = 0; nt < 2; ++nt)
#pragma unroll
        for (int mt = 0; mt < 2; ++mt)
#pragma unroll
          for (int i = 0; i < 8; ++i) {
            rz[g][nt][mt].f[i] = 0.f;
            if (g == 0) { inn[nt][mt].f[i] = 0.f; hnn[nt][mt].f[i] = 0.f; }
          }
    for (int kk = 0; kk < 8; ++kk) {  // A = new h0, B = w_ih1
      const int koff = kk * 32;
      FragA a0, a1;
      LOAD_A(a0, h0s, 0, koff);
      LOAD_A(a1, h0s, 1, koff);
#pragma unroll
      for (int g = 0; g < 3; ++g)
#pragma unroll
        for (int nt = 0; nt < 2; ++nt) {
          const int j = g * H_ + wave * 32 + nt * 16 + laneLo;
          FragA b;
          LOAD_B(b, wih1b, j, koff);
          if (g < 2) {
            rz[g][nt][0].v = WMMA(a0.v, b.v, rz[g][nt][0].v);
            rz[g][nt][1].v = WMMA(a1.v, b.v, rz[g][nt][1].v);
          } else {
            inn[nt][0].v = WMMA(a0.v, b.v, inn[nt][0].v);
            inn[nt][1].v = WMMA(a1.v, b.v, inn[nt][1].v);
          }
        }
    }
    for (int kk = 0; kk < 8; ++kk) {  // A = old h1, B = w_hh1
      const int koff = kk * 32;
      FragA a0, a1;
      LOAD_A(a0, h1s, 0, koff);
      LOAD_A(a1, h1s, 1, koff);
#pragma unroll
      for (int g = 0; g < 3; ++g)
#pragma unroll
        for (int nt = 0; nt < 2; ++nt) {
          const int j = g * H_ + wave * 32 + nt * 16 + laneLo;
          FragA b;
          LOAD_B(b, whh1b, j, koff);
          if (g < 2) {
            rz[g][nt][0].v = WMMA(a0.v, b.v, rz[g][nt][0].v);
            rz[g][nt][1].v = WMMA(a1.v, b.v, rz[g][nt][1].v);
          } else {
            hnn[nt][0].v = WMMA(a0.v, b.v, hnn[nt][0].v);
            hnn[nt][1].v = WMMA(a1.v, b.v, hnn[nt][1].v);
          }
        }
    }
#pragma unroll
    for (int nt = 0; nt < 2; ++nt)
#pragma unroll
      for (int mt = 0; mt < 2; ++mt)
#pragma unroll
        for (int r = 0; r < 8; ++r) {
          const float rg = sigmoidf_(rz[0][nt][mt].f[r] + br1[nt]);
          const float zg = sigmoidf_(rz[1][nt][mt].f[r] + bz1[nt]);
          const float ng = tanhf(inn[nt][mt].f[r] + bin1[nt] + rg * (hnn[nt][mt].f[r] + bhn1[nt]));
          h1r[nt][mt].f[r] = (1.f - zg) * ng + zg * h1r[nt][mt].f[r];
        }
    __syncthreads();  // done reading old h1s
#pragma unroll
    for (int nt = 0; nt < 2; ++nt)
#pragma unroll
      for (int mt = 0; mt < 2; ++mt)
#pragma unroll
        for (int r = 0; r < 8; ++r) {
          const int m = mt * 16 + r + 8 * laneHi;
          h1s[m * LDSW + wave * 32 + nt * 16 + laneLo] = f2bf(h1r[nt][mt].f[r]);
        }
    __syncthreads();  // new h1s visible for projections

    // ------- projections: pred(2) + chol z(3), 160 LDS-only dot tasks -------
    if (tid < ROWS * 5) {
      float a0 = 0.f, a1 = 0.f;
      const uint4* hp = (const uint4*)&h1s[prow * LDSW];
      const float* wv = &wps[po * H_];
      for (int k = 0; k < H_; k += 8) {
        const uint4 hq = hp[k >> 3];
        const float4 w0 = *(const float4*)(wv + k);
        const float4 w1 = *(const float4*)(wv + k + 4);
        a0 = fmaf(bflo(hq.x), w0.x, a0); a0 = fmaf(bfhi(hq.x), w0.y, a0);
        a0 = fmaf(bflo(hq.y), w0.z, a0); a0 = fmaf(bfhi(hq.y), w0.w, a0);
        a1 = fmaf(bflo(hq.z), w1.x, a1); a1 = fmaf(bfhi(hq.z), w1.y, a1);
        a1 = fmaf(bflo(hq.w), w1.z, a1); a1 = fmaf(bfhi(hq.w), w1.w, a1);
      }
      const float v = pbias + a0 + a1;
      const size_t b = row0 + prow;
      if (po == 0) {
        out[b * PL * 2 + t * 2] = v; poss[prow * 2] = v;
        out[outCh + b * PL * 4 + t * 4 + 1] = 0.f;
      } else if (po == 1) {
        out[b * PL * 2 + t * 2 + 1] = v; poss[prow * 2 + 1] = v;
      } else if (po == 2) {
        out[outCh + b * PL * 4 + t * 4 + 0] = softplusf_(v) + 1e-6f;
      } else if (po == 3) {
        out[outCh + b * PL * 4 + t * 4 + 2] = v;
      } else {
        out[outCh + b * PL * 4 + t * 4 + 3] = softplusf_(v) + 1e-6f;
      }
    }
    __syncthreads();  // pos/h1s settled before next step
  }
}

// ---------------------------------------------------------------------------
extern "C" void kernel_launch(void* const* d_in, const int* in_sizes, int n_in,
                              void* d_out, int out_size, void* d_ws, size_t ws_size,
                              hipStream_t stream) {
  const float* ctx    = (const float*)d_in[0];
  const int*   plen   = (const int*)d_in[1];
  const float* w_init = (const float*)d_in[2];
  const float* b_init = (const float*)d_in[3];
  const float* w_ih0  = (const float*)d_in[4];
  const float* w_hh0  = (const float*)d_in[5];
  const float* b_ih0  = (const float*)d_in[6];
  const float* b_hh0  = (const float*)d_in[7];
  const float* w_ih1  = (const float*)d_in[8];
  const float* w_hh1  = (const float*)d_in[9];
  const float* b_ih1  = (const float*)d_in[10];
  const float* b_hh1  = (const float*)d_in[11];
  const float* w_pos  = (const float*)d_in[12];
  const float* b_pos  = (const float*)d_in[13];
  const float* w_chol = (const float*)d_in[14];
  const float* b_chol = (const float*)d_in[15];
  float* out = (float*)d_out;

  const int Btot = in_sizes[0] / CDIM;  // 32768

  // workspace carve (all sizes 16B-aligned)
  char* w = (char*)d_ws;
  unsigned short* ctxb   = (unsigned short*)w; w += (size_t)Btot * CDIM * 2;
  unsigned short* winitb = (unsigned short*)w; w += (size_t)CDIM * CDIM * 2;
  unsigned short* wih0cb = (unsigned short*)w; w += (size_t)G3 * CDIM * 2;
  unsigned short* whh0b  = (unsigned short*)w; w += (size_t)G3 * H_ * 2;
  unsigned short* wih1b  = (unsigned short*)w; w += (size_t)G3 * H_ * 2;
  unsigned short* whh1b  = (unsigned short*)w; w += (size_t)G3 * H_ * 2;
  float* h0f  = (float*)w; w += (size_t)Btot * H_ * 4;
  float* h1f  = (float*)w; w += (size_t)Btot * H_ * 4;
  float* gi0f = (float*)w; w += (size_t)Btot * G3 * 4;

  // 1) bf16 conversions
  cvt_bf16<<<4096, 256, 0, stream>>>(ctx, ctxb, Btot * CDIM);
  cvt_bf16<<<256, 256, 0, stream>>>(w_init, winitb, CDIM * CDIM);
  cvt_bf16_strided<<<256, 256, 0, stream>>>(w_ih0, wih0cb, G3, CDIM, CDIM + 2, 2);
  cvt_bf16<<<256, 256, 0, stream>>>(w_hh0, whh0b, G3 * H_);
  cvt_bf16<<<256, 256, 0, stream>>>(w_ih1, wih1b, G3 * H_);
  cvt_bf16<<<256, 256, 0, stream>>>(w_hh1, whh1b, G3 * H_);

  // 2) fused init GEMM: N = 512 (hid, tanh) + 768 (gi0_ctx + biases) = 1280
  dim3 g1(Btot / 64, (CDIM + G3) / 64);
  init_gemm<<<g1, 256, 0, stream>>>(ctxb, winitb, wih0cb, b_init, b_ih0, b_hh0,
                                    h0f, h1f, gi0f);

  // 3) persistent decoder: 32 rows/block, all steps, gi0 LDS-resident
  decode<<<Btot / ROWS, 256, 0, stream>>>(whh0b, wih1b, whh1b, gi0f, w_ih0,
                                          b_hh0, b_ih1, b_hh1, h0f, h1f,
                                          w_pos, b_pos, w_chol, b_chol,
                                          plen, out, Btot);
}
